// ViTLayer_8117488190194
// MI455X (gfx1250) — 8-bit matrix operand order repaired and checked by reading, compiled, not run on hardware
//
#include <hip/hip_runtime.h>
#include <cstdint>
#include <cstddef>


typedef int v8i __attribute__((ext_vector_type(8)));

static constexpr int Bb   = 64;
static constexpr int Nt   = 197;
static constexpr int Hd   = 768;
static constexpr int Id   = 3072;
static constexpr int NH   = 12;
static constexpr int DH   = 64;
static constexpr int Mrows = Bb * Nt;
static constexpr int BH   = Bb * NH;
static constexpr int KP   = 256;

__device__ inline v8i load_a_frag(const int8_t* __restrict__ A, int row_base,
                                  int Mr, int K, int k0, int lane) {
  int m = row_base + (lane & 15);
  if (m >= Mr) m = Mr - 1;
  const uint8_t* p = (const uint8_t*)A + (size_t)m * K + k0;
  int h = (lane >> 4) & 1;
  v8i a = {0,0,0,0,0,0,0,0};
#pragma unroll
  for (int j = 0; j < 8; ++j) {
    int koff = 16 * (j >> 1) + 8 * h + 4 * (j & 1);
    a[j] = *(const int*)(p + koff);
  }
  return a;
}

__device__ inline v8i load_b_frag(const int8_t* __restrict__ Bp, int col_base,
                                  int Nc, int K, int k0, int lane) {
  int n = col_base + (lane & 15);
  if (n >= Nc) n = Nc - 1;
  const uint8_t* p = (const uint8_t*)Bp + (size_t)n * K + k0;
  int h = (lane >> 4) & 1;
  v8i b = {0,0,0,0,0,0,0,0};
#pragma unroll
  for (int j = 0; j < 8; ++j) {
    int koff = 16 * (j >> 1) + 8 * h + 4 * (j & 1);
    b[j] = *(const int*)(p + koff);
  }
  return b;
}

template <bool SA, bool SB>
__global__ void __launch_bounds__(256)
gemm_bin(const int8_t* __restrict__ A, const int8_t* __restrict__ Bp,
         float* __restrict__ C, const float* __restrict__ scale,
         const float* __restrict__ bias, int Mr, int O, int K, float smul) {
  const int lane = threadIdx.x & 31;
  const int wave = threadIdx.x >> 5;
  const int wm = wave & 3, wn = wave >> 2;
  const int m0 = blockIdx.x * 128 + wm * 32;
  const int c0 = blockIdx.y * 64 + wn * 32;

  v8i acc[2][2];
#pragma unroll
  for (int i = 0; i < 2; ++i)
#pragma unroll
    for (int j = 0; j < 2; ++j) acc[i][j] = (v8i){0,0,0,0,0,0,0,0};

  for (int k0 = 0; k0 < K; k0 += 64) {
    v8i a0 = load_a_frag(A, m0,      Mr, K, k0, lane);
    v8i a1 = load_a_frag(A, m0 + 16, Mr, K, k0, lane);
    v8i b0 = load_b_frag(Bp, c0,      O, K, k0, lane);
    v8i b1 = load_b_frag(Bp, c0 + 16, O, K, k0, lane);
    acc[0][0] = __builtin_amdgcn_wmma_i32_16x16x64_iu8(SA, a0, SB, b0, acc[0][0], false, false);
    acc[0][1] = __builtin_amdgcn_wmma_i32_16x16x64_iu8(SA, a0, SB, b1, acc[0][1], false, false);
    acc[1][0] = __builtin_amdgcn_wmma_i32_16x16x64_iu8(SA, a1, SB, b0, acc[1][0], false, false);
    acc[1][1] = __builtin_amdgcn_wmma_i32_16x16x64_iu8(SA, a1, SB, b1, acc[1][1], false, false);
  }

  const int h = (lane >> 4) & 1;
  const int nloc = lane & 15;
#pragma unroll
  for (int mt = 0; mt < 2; ++mt)
#pragma unroll
    for (int nt = 0; nt < 2; ++nt) {
      int col = c0 + nt * 16 + nloc;
      float sc = (scale ? scale[col] : 1.0f) * smul;
      float bi = bias ? bias[col] : 0.0f;
#pragma unroll
      for (int r = 0; r < 8; ++r) {
        int row = m0 + mt * 16 + r + 8 * h;
        if (row < Mr) C[(size_t)row * O + col] = (float)acc[mt][nt][r] * sc + bi;
      }
    }
}

__global__ void __launch_bounds__(256)
attn_scores(const int8_t* __restrict__ Qp, const int8_t* __restrict__ Kp,
            float* __restrict__ S) {
  const int lane = threadIdx.x & 31;
  const int wave = threadIdx.x >> 5;
  const int bh = blockIdx.z;
  const int mt = blockIdx.x;
  const int nt = blockIdx.y * 8 + wave;
  if (nt > 12) return;
  const int8_t* q = Qp + (size_t)bh * Nt * DH;
  const int8_t* k = Kp + (size_t)bh * Nt * DH;
  v8i a = load_a_frag(q, mt * 16, Nt, DH, 0, lane);
  v8i b = load_b_frag(k, nt * 16, Nt, DH, 0, lane);
  v8i acc = (v8i){0,0,0,0,0,0,0,0};
  acc = __builtin_amdgcn_wmma_i32_16x16x64_iu8(true, a, true, b, acc, false, false);
  const int h = (lane >> 4) & 1;
  const int n = nt * 16 + (lane & 15);
  float* s = S + (size_t)bh * Nt * Nt;
#pragma unroll
  for (int r = 0; r < 8; ++r) {
    int m = mt * 16 + r + 8 * h;
    if (m < Nt && n < Nt) s[(size_t)m * Nt + n] = (float)acc[r] * 0.125f;
  }
}

__global__ void __launch_bounds__(256)
softmax_quant(const float* __restrict__ S, uint8_t* __restrict__ bits,
              const float* __restrict__ alpha_p) {
  const int lane = threadIdx.x & 31;
  const int wave = threadIdx.x >> 5;
  const long row = (long)blockIdx.x * 8 + wave;
  if (row >= (long)BH * Nt) return;
  const float* s = S + row * Nt;
  const float alpha = alpha_p[0];
  float vals[8];
  float mx = -3.0e38f;
#pragma unroll
  for (int i = 0; i < 8; ++i) {
    int j = lane + i * 32;
    vals[i] = (j < Nt) ? s[j] : -3.0e38f;
    mx = fmaxf(mx, vals[i]);
  }
  for (int off = 16; off; off >>= 1) mx = fmaxf(mx, __shfl_xor(mx, off, 32));
  float sum = 0.0f;
#pragma unroll
  for (int i = 0; i < 8; ++i) {
    vals[i] = (vals[i] > -1.0e38f) ? __expf(vals[i] - mx) : 0.0f;
    sum += vals[i];
  }
  for (int off = 16; off; off >>= 1) sum += __shfl_xor(sum, off, 32);
  const float inv = 1.0f / sum;
  uint8_t* o = bits + row * KP;
#pragma unroll
  for (int i = 0; i < 8; ++i) {
    int j = lane + i * 32;
    if (j < KP) {
      uint8_t v = 0;
      if (j < Nt) {
        float p = vals[i] * inv;
        float q = rintf(p / alpha);
        v = (uint8_t)fminf(fmaxf(q, 0.0f), 1.0f);
      }
      o[j] = v;
    }
  }
}

__global__ void __launch_bounds__(128)
attn_ctx(const uint8_t* __restrict__ bits, const int8_t* __restrict__ vT,
         float* __restrict__ ctx, const float* __restrict__ alpha_p) {
  const int lane = threadIdx.x & 31;
  const int wave = threadIdx.x >> 5;
  const int bh = blockIdx.y;
  const int mt = blockIdx.x;
  const int b = bh / NH, hh = bh % NH;
  const int8_t* A  = (const int8_t*)bits + (size_t)bh * Nt * KP;
  const int8_t* Bv = vT + (size_t)bh * DH * KP;
  v8i acc = (v8i){0,0,0,0,0,0,0,0};
  for (int k0 = 0; k0 < KP; k0 += 64) {
    v8i a = load_a_frag(A, mt * 16, Nt, KP, k0, lane);
    v8i v = load_b_frag(Bv, wave * 16, DH, KP, k0, lane);
    acc = __builtin_amdgcn_wmma_i32_16x16x64_iu8(false, a, true, v, acc, false, false);
  }
  const float alpha = alpha_p[0];
  const int h = (lane >> 4) & 1;
  const int d = wave * 16 + (lane & 15);
#pragma unroll
  for (int r = 0; r < 8; ++r) {
    int m = mt * 16 + r + 8 * h;
    if (m < Nt)
      ctx[((size_t)(b * Nt + m)) * Hd + hh * DH + d] = (float)acc[r] * alpha;
  }
}

__device__ inline void block_meanvar(const float* __restrict__ x, int D,
                                     float& mean, float& rstd, float* sm) {
  float s = 0.0f, s2 = 0.0f;
  for (int j = threadIdx.x; j < D; j += blockDim.x) {
    float v = x[j];
    s += v; s2 += v * v;
  }
  for (int off = 16; off; off >>= 1) {
    s  += __shfl_xor(s,  off, 32);
    s2 += __shfl_xor(s2, off, 32);
  }
  const int wave = threadIdx.x >> 5;
  if ((threadIdx.x & 31) == 0) { sm[wave] = s; sm[wave + 8] = s2; }
  __syncthreads();
  if (threadIdx.x < 32) {
    float a = (threadIdx.x < 8) ? sm[threadIdx.x] : 0.0f;
    float b = (threadIdx.x < 8) ? sm[threadIdx.x + 8] : 0.0f;
    for (int off = 4; off; off >>= 1) {
      a += __shfl_xor(a, off, 32);
      b += __shfl_xor(b, off, 32);
    }
    if (threadIdx.x == 0) { sm[16] = a; sm[17] = b; }
  }
  __syncthreads();
  float mu  = sm[16] / (float)D;
  float var = sm[17] / (float)D - mu * mu;
  mean = mu;
  rstd = rsqrtf(var + 1e-6f);
}

__global__ void __launch_bounds__(256)
ln_plain(const float* __restrict__ X, float* __restrict__ Y,
         const float* __restrict__ g, const float* __restrict__ b, int D) {
  __shared__ float sm[32];
  const float* x = X + (size_t)blockIdx.x * D;
  float mean, rstd;
  block_meanvar(x, D, mean, rstd, sm);
  float* y = Y + (size_t)blockIdx.x * D;
  for (int j = threadIdx.x; j < D; j += blockDim.x)
    y[j] = (x[j] - mean) * rstd * g[j] + b[j];
}

__global__ void __launch_bounds__(256)
pack_sign(const float* __restrict__ X, const float* __restrict__ mv,
          int8_t* __restrict__ P, int D) {
  const size_t base = (size_t)blockIdx.x * D;
  for (int j = threadIdx.x; j < D; j += blockDim.x)
    P[base + j] = (X[base + j] + mv[j] >= 0.0f) ? (int8_t)1 : (int8_t)-1;
}

__global__ void __launch_bounds__(256)
pack_w(const float* __restrict__ W, int8_t* __restrict__ P,
       float* __restrict__ scale, int Kd) {
  __shared__ float sm[32];
  const size_t base = (size_t)blockIdx.x * Kd;
  float s = 0.0f;
  for (int j = threadIdx.x; j < Kd; j += blockDim.x) {
    float w = W[base + j];
    s += fabsf(w);
    P[base + j] = (w >= 0.0f) ? (int8_t)1 : (int8_t)-1;
  }
  for (int off = 16; off; off >>= 1) s += __shfl_xor(s, off, 32);
  const int wave = threadIdx.x >> 5;
  if ((threadIdx.x & 31) == 0) sm[wave] = s;
  __syncthreads();
  if (threadIdx.x == 0) {
    float a = 0.0f;
    for (int i = 0; i < 8; ++i) a += sm[i];
    scale[blockIdx.x] = a / (float)Kd;
  }
}

__global__ void __launch_bounds__(256)
qkv_epi(const float* __restrict__ C, const float* __restrict__ HX,
        const float* __restrict__ g, const float* __restrict__ bb,
        const float* __restrict__ mv1, const float* __restrict__ pre,
        const float* __restrict__ mv2, const float* __restrict__ move2,
        float* __restrict__ Mout, int8_t* __restrict__ pack, int transposed) {
  __shared__ float sm[32];
  const size_t row = blockIdx.x;
  const float* c = C + row * Hd;
  float mean, rstd;
  block_meanvar(c, Hd, mean, rstd, sm);
  const int b_ = (int)(row / Nt), t = (int)(row % Nt);
  for (int j = threadIdx.x; j < Hd; j += blockDim.x) {
    float y = (c[j] - mean) * rstd * g[j] + bb[j] + HX[row * Hd + j];
    float z = y - mv1[j];
    z = (z >= 0.0f ? z : z * pre[j]) + mv2[j];
    Mout[row * Hd + j] = z;
    int8_t sg = (z + move2[j] >= 0.0f) ? (int8_t)1 : (int8_t)-1;
    int hh = j >> 6, d = j & 63;
    size_t idx = transposed
        ? ((size_t)((b_ * NH + hh) * DH + d) * KP + t)
        : ((size_t)((b_ * NH + hh) * Nt + t) * DH + d);
    pack[idx] = sg;
  }
}

__global__ void __launch_bounds__(256)
ctx_epi(const float* __restrict__ C, const float* __restrict__ Mq,
        const float* __restrict__ Mk, const float* __restrict__ Mv,
        const float* __restrict__ g, const float* __restrict__ bb,
        const float* __restrict__ mv1, const float* __restrict__ pre,
        const float* __restrict__ mv2, float* __restrict__ Out) {
  __shared__ float sm[32];
  const size_t row = blockIdx.x;
  const float* c = C + row * Hd;
  float mean, rstd;
  block_meanvar(c, Hd, mean, rstd, sm);
  for (int j = threadIdx.x; j < Hd; j += blockDim.x) {
    size_t k = row * Hd + j;
    float y = (c[j] - mean) * rstd * g[j] + bb[j] + Mq[k] + Mk[k] + Mv[k];
    float z = y - mv1[j];
    Out[k] = (z >= 0.0f ? z : z * pre[j]) + mv2[j];
  }
}

__global__ void __launch_bounds__(256)
selfout_epi(const float* __restrict__ C, const float* __restrict__ A1,
            const float* __restrict__ X0, const float* __restrict__ g,
            const float* __restrict__ bb, const float* __restrict__ mv1,
            const float* __restrict__ pre, const float* __restrict__ mv2,
            const float* __restrict__ lsA, const float* __restrict__ lsM,
            float* __restrict__ X1) {
  __shared__ float sm[32];
  const size_t row = blockIdx.x;
  const float* c = C + row * Hd;
  float mean, rstd;
  block_meanvar(c, Hd, mean, rstd, sm);
  for (int j = threadIdx.x; j < Hd; j += blockDim.x) {
    size_t k = row * Hd + j;
    float y = (c[j] - mean) * rstd * g[j] + bb[j] + A1[k];
    float z = y - mv1[j];
    z = (z >= 0.0f ? z : z * pre[j]) + mv2[j];
    X1[k] = z * lsA[j] + lsM[j] + X0[k];
  }
}

__global__ void __launch_bounds__(256)
inter_epi(float* __restrict__ C, const float* __restrict__ HY,
          const float* __restrict__ g, const float* __restrict__ bb,
          const float* __restrict__ mv1, const float* __restrict__ pre,
          const float* __restrict__ mv2) {
  __shared__ float sm[32];
  const size_t row = blockIdx.x;
  float* c = C + row * Id;
  float mean, rstd;
  block_meanvar(c, Id, mean, rstd, sm);
  for (int j = threadIdx.x; j < Id; j += blockDim.x) {
    float y = (c[j] - mean) * rstd * g[j] + bb[j] + HY[row * Hd + (j % Hd)];
    float z = y - mv1[j];
    c[j] = (z >= 0.0f ? z : z * pre[j]) + mv2[j];
  }
}

__global__ void __launch_bounds__(256)
out_epi(const float* __restrict__ C, const float* __restrict__ Y,
        const float* __restrict__ X1, const float* __restrict__ g,
        const float* __restrict__ bb, const float* __restrict__ mv1,
        const float* __restrict__ pre, const float* __restrict__ mv2,
        const float* __restrict__ lsA, const float* __restrict__ lsM,
        float* __restrict__ Out) {
  __shared__ float sm[32];
  const size_t row = blockIdx.x;
  const float* c = C + row * Hd;
  float mean, rstd;
  block_meanvar(c, Hd, mean, rstd, sm);
  const float* y4 = Y + row * Id;
  for (int j = threadIdx.x; j < Hd; j += blockDim.x) {
    float pooled = 0.25f * (y4[4 * j] + y4[4 * j + 1] + y4[4 * j + 2] + y4[4 * j + 3]);
    float y = (c[j] - mean) * rstd * g[j] + bb[j] + pooled;
    float z = y - mv1[j];
    z = (z >= 0.0f ? z : z * pre[j]) + mv2[j];
    Out[row * Hd + j] = z * lsA[j] + lsM[j] + X1[row * Hd + j];
  }
}

extern "C" void kernel_launch(void* const* d_in, const int* in_sizes, int n_in,
                              void* d_out, int out_size, void* d_ws, size_t ws_size,
                              hipStream_t stream) {
  (void)in_sizes; (void)n_in; (void)out_size; (void)ws_size;
  const float* F = nullptr; (void)F;

  const float* x        = (const float*)d_in[0];
  const float* ln0_g    = (const float*)d_in[1];
  const float* ln0_b    = (const float*)d_in[2];
  const float* lnA_g    = (const float*)d_in[3];
  const float* lnA_b    = (const float*)d_in[4];
  const float* moveq    = (const float*)d_in[5];
  const float* movek    = (const float*)d_in[6];
  const float* movev    = (const float*)d_in[7];
  const float* Wq       = (const float*)d_in[8];
  const float* bq       = (const float*)d_in[9];
  const float* Wk       = (const float*)d_in[10];
  const float* bk       = (const float*)d_in[11];
  const float* Wv       = (const float*)d_in[12];
  const float* bv       = (const float*)d_in[13];
  const float* nq_g     = (const float*)d_in[14];
  const float* nq_b     = (const float*)d_in[15];
  const float* nk_g     = (const float*)d_in[16];
  const float* nk_b     = (const float*)d_in[17];
  const float* nv_g     = (const float*)d_in[18];
  const float* nv_b     = (const float*)d_in[19];
  const float* rq1      = (const float*)d_in[20];
  const float* rqp      = (const float*)d_in[21];
  const float* rq2      = (const float*)d_in[22];
  const float* rk1      = (const float*)d_in[23];
  const float* rkp      = (const float*)d_in[24];
  const float* rk2      = (const float*)d_in[25];
  const float* rv1      = (const float*)d_in[26];
  const float* rvp      = (const float*)d_in[27];
  const float* rv2      = (const float*)d_in[28];
  const float* moveq2   = (const float*)d_in[29];
  const float* movek2   = (const float*)d_in[30];
  const float* movev2   = (const float*)d_in[31];
  const float* alpha    = (const float*)d_in[32];
  const float* nc_g     = (const float*)d_in[33];
  const float* nc_b     = (const float*)d_in[34];
  const float* rc1      = (const float*)d_in[35];
  const float* rcp      = (const float*)d_in[36];
  const float* rc2      = (const float*)d_in[37];
  const float* Wd       = (const float*)d_in[38];
  const float* bd       = (const float*)d_in[39];
  const float* so_move  = (const float*)d_in[40];
  const float* so_g     = (const float*)d_in[41];
  const float* so_b     = (const float*)d_in[42];
  const float* so_r1    = (const float*)d_in[43];
  const float* so_rp    = (const float*)d_in[44];
  const float* so_r2    = (const float*)d_in[45];
  const float* so_lsA   = (const float*)d_in[46];
  const float* so_lsM   = (const float*)d_in[47];
  const float* Wi       = (const float*)d_in[48];
  const float* bi       = (const float*)d_in[49];
  const float* i_move   = (const float*)d_in[50];
  const float* i_g      = (const float*)d_in[51];
  const float* i_b      = (const float*)d_in[52];
  const float* i_r1     = (const float*)d_in[53];
  const float* i_rp     = (const float*)d_in[54];
  const float* i_r2     = (const float*)d_in[55];
  const float* Wo       = (const float*)d_in[56];
  const float* bo       = (const float*)d_in[57];
  const float* o_move   = (const float*)d_in[58];
  const float* o_g      = (const float*)d_in[59];
  const float* o_b      = (const float*)d_in[60];
  const float* o_r1     = (const float*)d_in[61];
  const float* o_rp     = (const float*)d_in[62];
  const float* o_r2     = (const float*)d_in[63];
  const float* o_lsA    = (const float*)d_in[64];
  const float* o_lsM    = (const float*)d_in[65];
  float* out            = (float*)d_out;

  char* Wp = (char*)d_ws;
  size_t off = 0;
  auto carve = [&](size_t bytes) -> void* {
    void* p = Wp + off;
    off += (bytes + 255) & ~(size_t)255;
    return p;
  };
  const size_t MHf = (size_t)Mrows * Hd * sizeof(float);
  const size_t MIf = (size_t)Mrows * Id * sizeof(float);

  float*  f_hx  = (float*)carve(MHf);
  float*  f_c   = (float*)carve(MHf);
  float*  f_mq  = (float*)carve(MHf);
  float*  f_mk  = (float*)carve(MHf);
  float*  f_mv  = (float*)carve(MHf);
  float*  f_a1  = (float*)carve(MHf);
  float*  f_x1  = (float*)carve(MHf);
  float*  f_ci  = (float*)carve(MIf);
  int8_t* i8_act = (int8_t*)carve((size_t)Mrows * Id);
  int8_t* i8_q  = (int8_t*)carve((size_t)BH * Nt * DH);
  int8_t* i8_k  = (int8_t*)carve((size_t)BH * Nt * DH);
  int8_t* i8_vT = (int8_t*)carve((size_t)BH * DH * KP);
  float*  f_sc  = (float*)carve((size_t)BH * Nt * Nt * sizeof(float));
  uint8_t* u8_p = (uint8_t*)carve((size_t)BH * Nt * KP);
  int8_t* w_q   = (int8_t*)carve((size_t)Hd * Hd);
  int8_t* w_k   = (int8_t*)carve((size_t)Hd * Hd);
  int8_t* w_v   = (int8_t*)carve((size_t)Hd * Hd);
  int8_t* w_d   = (int8_t*)carve((size_t)Hd * Hd);
  int8_t* w_i   = (int8_t*)carve((size_t)Id * Hd);
  int8_t* w_o   = (int8_t*)carve((size_t)Hd * Id);
  float*  s_q   = (float*)carve(Hd * sizeof(float));
  float*  s_k   = (float*)carve(Hd * sizeof(float));
  float*  s_v   = (float*)carve(Hd * sizeof(float));
  float*  s_d   = (float*)carve(Hd * sizeof(float));
  float*  s_i   = (float*)carve(Id * sizeof(float));
  float*  s_o   = (float*)carve(Hd * sizeof(float));

  const dim3 blk(256);
  const dim3 gW(99, 12);
  const dim3 gI(99, 48);

  pack_w<<<Hd, blk, 0, stream>>>(Wq, w_q, s_q, Hd);
  pack_w<<<Hd, blk, 0, stream>>>(Wk, w_k, s_k, Hd);
  pack_w<<<Hd, blk, 0, stream>>>(Wv, w_v, s_v, Hd);
  pack_w<<<Hd, blk, 0, stream>>>(Wd, w_d, s_d, Hd);
  pack_w<<<Id, blk, 0, stream>>>(Wi, w_i, s_i, Hd);
  pack_w<<<Hd, blk, 0, stream>>>(Wo, w_o, s_o, Id);

  ln_plain<<<Mrows, blk, 0, stream>>>(x, f_hx, ln0_g, ln0_b, Hd);

  pack_sign<<<Mrows, blk, 0, stream>>>(f_hx, moveq, i8_act, Hd);
  gemm_bin<true, true><<<gW, blk, 0, stream>>>(i8_act, w_q, f_c, s_q, bq, Mrows, Hd, Hd, 1.0f);
  qkv_epi<<<Mrows, blk, 0, stream>>>(f_c, f_hx, nq_g, nq_b, rq1, rqp, rq2, moveq2, f_mq, i8_q, 0);

  pack_sign<<<Mrows, blk, 0, stream>>>(f_hx, movek, i8_act, Hd);
  gemm_bin<true, true><<<gW, blk, 0, stream>>>(i8_act, w_k, f_c, s_k, bk, Mrows, Hd, Hd, 1.0f);
  qkv_epi<<<Mrows, blk, 0, stream>>>(f_c, f_hx, nk_g, nk_b, rk1, rkp, rk2, movek2, f_mk, i8_k, 0);

  pack_sign<<<Mrows, blk, 0, stream>>>(f_hx, movev, i8_act, Hd);
  gemm_bin<true, true><<<gW, blk, 0, stream>>>(i8_act, w_v, f_c, s_v, bv, Mrows, Hd, Hd, 1.0f);
  qkv_epi<<<Mrows, blk, 0, stream>>>(f_c, f_hx, nv_g, nv_b, rv1, rvp, rv2, movev2, f_mv, i8_vT, 1);

  attn_scores<<<dim3(13, 2, BH), blk, 0, stream>>>(i8_q, i8_k, f_sc);
  softmax_quant<<<(BH * Nt + 7) / 8, blk, 0, stream>>>(f_sc, u8_p, alpha);
  attn_ctx<<<dim3(13, BH), dim3(128), 0, stream>>>(u8_p, i8_vT, f_c, alpha);
  ctx_epi<<<Mrows, blk, 0, stream>>>(f_c, f_mq, f_mk, f_mv, nc_g, nc_b, rc1, rcp, rc2, f_a1);

  pack_sign<<<Mrows, blk, 0, stream>>>(f_a1, so_move, i8_act, Hd);
  gemm_bin<true, true><<<gW, blk, 0, stream>>>(i8_act, w_d, f_c, s_d, bd, Mrows, Hd, Hd, 1.0f);
  selfout_epi<<<Mrows, blk, 0, stream>>>(f_c, f_a1, x, so_g, so_b, so_r1, so_rp, so_r2,
                                         so_lsA, so_lsM, f_x1);

  ln_plain<<<Mrows, blk, 0, stream>>>(f_x1, f_hx, lnA_g, lnA_b, Hd);
  pack_sign<<<Mrows, blk, 0, stream>>>(f_hx, i_move, i8_act, Hd);
  gemm_bin<true, true><<<gI, blk, 0, stream>>>(i8_act, w_i, f_ci, s_i, bi, Mrows, Id, Hd, 1.0f);
  inter_epi<<<Mrows, blk, 0, stream>>>(f_ci, f_hx, i_g, i_b, i_r1, i_rp, i_r2);

  pack_sign<<<Mrows, blk, 0, stream>>>(f_ci, o_move, i8_act, Id);
  gemm_bin<true, true><<<gW, blk, 0, stream>>>(i8_act, w_o, f_c, s_o, bo, Mrows, Hd, Id, 1.0f);
  out_epi<<<Mrows, blk, 0, stream>>>(f_c, f_ci, f_x1, o_g, o_b, o_r1, o_rp, o_r2,
                                     o_lsA, o_lsM, out);
}
